// BasicRNN_12292196401858
// MI455X (gfx1250) — matrix operand order repaired and checked by reading, compiled, not run on hardware
//
#include <hip/hip_runtime.h>

// ---------------------------------------------------------------------------
// SimpleRNN  (B=128, T=512, D=U=1024)  for gfx1250 (CDNA5, wave32, WMMA)
//   states[t] = tanh(X[:,t,:] @ W_xh + b + h_{t-1} @ W_hh)
// bf16 operands, f32 accumulation via v_wmma_f32_16x16x32_bf16.
// Global->LDS staging: CDNA5 async copies (ASYNCcnt), double-buffered.
// Recurrent scan: ONE persistent kernel (software grid barrier), no per-step
// launch overhead.
// ---------------------------------------------------------------------------

typedef __attribute__((ext_vector_type(16))) __bf16 v16bf;
typedef __attribute__((ext_vector_type(8)))  float  v8f;

struct Frag16 { uint4 lo; uint4 hi; };   // 32 bytes == v16bf

static constexpr int Bsz = 128;
static constexpr int T   = 512;
static constexpr int D   = 1024;
static constexpr int U   = 1024;

__device__ __forceinline__ unsigned short f32_to_bf16(float f) {
    unsigned int u = __float_as_uint(f);
    unsigned int r = u + 0x7FFFu + ((u >> 16) & 1u);   // round-to-nearest-even
    return (unsigned short)(r >> 16);
}

// --- CDNA5 async global->LDS copy (16B), tracked by ASYNCcnt ---------------
__device__ __forceinline__ void async_copy_b128(unsigned short* lds_dst,
                                                const unsigned short* gsrc) {
    unsigned lds_off = (unsigned)(unsigned long long)(uintptr_t)lds_dst;
    unsigned long long ga = (unsigned long long)(uintptr_t)gsrc;
    asm volatile("global_load_async_to_lds_b128 %0, %1, off"
                 :: "v"(lds_off), "v"(ga) : "memory");
}
template <int N>
__device__ __forceinline__ void wait_async() {
    asm volatile("s_wait_asynccnt %0" :: "i"(N) : "memory");
}

__device__ __forceinline__ v16bf load_frag(const unsigned short* p0,
                                           const unsigned short* p1) {
    Frag16 f;
    f.lo = *reinterpret_cast<const uint4*>(p0);
    f.hi = *reinterpret_cast<const uint4*>(p1);
    return __builtin_bit_cast(v16bf, f);
}

// ---------------------------------------------------------------------------
// prep: W_xh -> bf16 W_xh^T, W_hh -> bf16 W_hh^T (N-major), zero h0 + sync
// ---------------------------------------------------------------------------
__global__ __launch_bounds__(256) void prep(const float* __restrict__ Wxh,
                                            const float* __restrict__ Whh,
                                            unsigned short* __restrict__ wxhT,
                                            unsigned short* __restrict__ whhT,
                                            unsigned short* __restrict__ h0,
                                            unsigned* __restrict__ syncw) {
    const int NW = D * U;
    int idx = blockIdx.x * 256 + threadIdx.x;
    if (idx < NW) {
        int n = idx >> 10, k = idx & 1023;
        wxhT[idx] = f32_to_bf16(Wxh[(size_t)k * U + n]);
    } else if (idx < 2 * NW) {
        int j = idx - NW;
        int n = j >> 10, k = j & 1023;
        whhT[j] = f32_to_bf16(Whh[(size_t)k * U + n]);
    } else if (idx < 2 * NW + Bsz * U) {
        h0[idx - 2 * NW] = 0;
    } else if (idx < 2 * NW + Bsz * U + 16) {
        syncw[idx - 2 * NW - Bsz * U] = 0u;    // grid-barrier counters
    }
}

// ---------------------------------------------------------------------------
// xcvt: X fp32 -> bf16 (row-major), 16 elems/thread
// ---------------------------------------------------------------------------
__global__ __launch_bounds__(256) void xcvt(const float* __restrict__ X,
                                            unsigned short* __restrict__ xbf) {
    size_t i = ((size_t)blockIdx.x * 256 + threadIdx.x) * 16;
    unsigned short tmp[16];
    #pragma unroll
    for (int q = 0; q < 4; ++q) {
        float4 v = reinterpret_cast<const float4*>(X + i)[q];
        tmp[q * 4 + 0] = f32_to_bf16(v.x);
        tmp[q * 4 + 1] = f32_to_bf16(v.y);
        tmp[q * 4 + 2] = f32_to_bf16(v.z);
        tmp[q * 4 + 3] = f32_to_bf16(v.w);
    }
    #pragma unroll
    for (int q = 0; q < 2; ++q)
        reinterpret_cast<uint4*>(xbf + i)[q] =
            reinterpret_cast<const uint4*>(tmp)[q];
}

// ---------------------------------------------------------------------------
// compute core per K-chunk: wave owns 16(M) x 128(N) = 8 wmma tiles
// ---------------------------------------------------------------------------
__device__ __forceinline__ void wmma_strip(const unsigned short* sA,
                                           const unsigned short* sB,
                                           int wave, int lr, int kbA, int kbB,
                                           v8f acc[8]) {
    const unsigned short* arow = &sA[(wave * 16 + lr) * 32];
    v16bf va = load_frag(arow + kbB, arow + kbB + 8);
    const unsigned short* b0 = &sB[lr * 32 + kbB];
    v16bf vb = load_frag(b0, b0 + 8);
    #pragma unroll
    for (int nt = 0; nt < 8; ++nt) {
        v16bf vb_cur = vb;
        if (nt < 7) {
            const unsigned short* bn = &sB[((nt + 1) * 16 + lr) * 32 + kbB];
            vb = load_frag(bn, bn + 8);
        }
        acc[nt] = __builtin_amdgcn_wmma_f32_16x16x32_bf16(
            false, va, false, vb_cur, (short)0, acc[nt], false, false);
    }
}

// --- software grid barrier (agent scope) -----------------------------------
__device__ __forceinline__ void grid_sync(unsigned* counter, unsigned* gen,
                                          unsigned nblocks, unsigned target) {
    __syncthreads();
    if (threadIdx.x == 0) {
        __threadfence();   // publish this block's global writes (agent scope)
        unsigned old = __hip_atomic_fetch_add(counter, 1u, __ATOMIC_ACQ_REL,
                                              __HIP_MEMORY_SCOPE_AGENT);
        if (old == nblocks - 1) {
            __hip_atomic_store(counter, 0u, __ATOMIC_RELAXED,
                               __HIP_MEMORY_SCOPE_AGENT);
            __hip_atomic_fetch_add(gen, 1u, __ATOMIC_ACQ_REL,
                                   __HIP_MEMORY_SCOPE_AGENT);
        } else {
            while (__hip_atomic_load(gen, __ATOMIC_ACQUIRE,
                                     __HIP_MEMORY_SCOPE_AGENT) < target) {
                __builtin_amdgcn_s_sleep(1);
            }
        }
    }
    __syncthreads();
}

// ---------------------------------------------------------------------------
// xproj: out[t*B*U + b*U + u] = X[b*T+t,:] @ W_xh[:,u] + bias[u]
// PRE=true: bf16 X, double-buffered async staging for A and B.
// ---------------------------------------------------------------------------
template <bool PRE>
__global__ __launch_bounds__(256) void xproj(const float* __restrict__ X,
                                             const unsigned short* __restrict__ Xbf,
                                             const unsigned short* __restrict__ WT,
                                             const float* __restrict__ bias,
                                             float* __restrict__ out) {
    __shared__ __align__(16) unsigned short sA[2][128 * 32];
    __shared__ __align__(16) unsigned short sB[2][128 * 32];

    const int n0 = blockIdx.x * 128;
    const int r0 = blockIdx.y * 128;

    const int tid  = threadIdx.x;
    const int lane = tid & 31;
    const int wave = tid >> 5;
    const int lr   = lane & 15;
    const int kbA  = (lane >> 4) << 3;    // 0 or 8
    const int kbB  = (lane >> 4) << 4;    // 0 or 16

    const int lrow  = tid >> 1;           // 0..127
    const int lhalf = (tid & 1) << 4;     // 0 or 16

    v8f acc[8];
    #pragma unroll
    for (int n = 0; n < 8; ++n)
        acc[n] = v8f{0.f, 0.f, 0.f, 0.f, 0.f, 0.f, 0.f, 0.f};

    if constexpr (PRE) {
        auto stage = [&](int buf, int kk) {
            const unsigned short* gB = WT + (size_t)(n0 + lrow) * D + kk + lhalf;
            unsigned short* dB = &sB[buf][lrow * 32 + lhalf];
            async_copy_b128(dB,     gB);
            async_copy_b128(dB + 8, gB + 8);
            const unsigned short* gA = Xbf + (size_t)(r0 + lrow) * D + kk + lhalf;
            unsigned short* dA = &sA[buf][lrow * 32 + lhalf];
            async_copy_b128(dA,     gA);
            async_copy_b128(dA + 8, gA + 8);
        };
        stage(0, 0);
        int cb = 0;
        for (int kk = 0; kk < D; kk += 32) {
            if (kk + 32 < D) {
                stage(cb ^ 1, kk + 32);   // stream next chunk while computing
                wait_async<4>();          // in-order: current chunk landed
            } else {
                wait_async<0>();
            }
            __syncthreads();
            wmma_strip(sA[cb], sB[cb], wave, lr, kbA, kbB, acc);
            __syncthreads();
            cb ^= 1;
        }
    } else {
        for (int kk = 0; kk < D; kk += 32) {
            const unsigned short* gB = WT + (size_t)(n0 + lrow) * D + kk + lhalf;
            unsigned short* dB = &sB[0][lrow * 32 + lhalf];
            async_copy_b128(dB,     gB);
            async_copy_b128(dB + 8, gB + 8);
            const float* gx = X + (size_t)(r0 + lrow) * D + kk + lhalf;
            unsigned short* dst = &sA[0][lrow * 32 + lhalf];
            #pragma unroll
            for (int q = 0; q < 4; ++q) {
                float4 v = reinterpret_cast<const float4*>(gx)[q];
                dst[q * 4 + 0] = f32_to_bf16(v.x);
                dst[q * 4 + 1] = f32_to_bf16(v.y);
                dst[q * 4 + 2] = f32_to_bf16(v.z);
                dst[q * 4 + 3] = f32_to_bf16(v.w);
            }
            wait_async<0>();
            __syncthreads();
            wmma_strip(sA[0], sB[0], wave, lr, kbA, kbB, acc);
            __syncthreads();
        }
    }

    // epilogue: +bias, scatter to [T,B,U]
    const int mbase = r0 + wave * 16 + ((lane >> 4) << 3);
    #pragma unroll
    for (int nt = 0; nt < 8; ++nt) {
        const int u = n0 + nt * 16 + lr;
        const float bv = bias[u];
        #pragma unroll
        for (int i = 0; i < 8; ++i) {
            const int r  = mbase + i;          // global row = b*T + t
            const int bb = r >> 9;
            const int tt = r & 511;
            out[((size_t)(tt * Bsz + bb)) * U + u] = acc[nt][i] + bv;
        }
    }
}

// ---------------------------------------------------------------------------
// rnn_scan: persistent kernel, all T steps.  grid = 8 blocks x 256.
//   per step: outT[m,u] = tanh(outT[m,u] + h_prev @ W_hh); hnext = bf16
//   blocks synchronize with a software grid barrier between steps.
// ---------------------------------------------------------------------------
__global__ __launch_bounds__(256) void rnn_scan(unsigned short* __restrict__ h0,
                                                unsigned short* __restrict__ h1,
                                                const unsigned short* __restrict__ WT,
                                                float* __restrict__ out,
                                                unsigned* __restrict__ counter,
                                                unsigned* __restrict__ gen) {
    __shared__ __align__(16) unsigned short sA[2][128 * 32];
    __shared__ __align__(16) unsigned short sB[2][128 * 32];

    const int n0 = blockIdx.x * 128;

    const int tid  = threadIdx.x;
    const int lane = tid & 31;
    const int wave = tid >> 5;
    const int lr   = lane & 15;
    const int kbA  = (lane >> 4) << 3;
    const int kbB  = (lane >> 4) << 4;

    const int lrow  = tid >> 1;
    const int lhalf = (tid & 1) << 4;
    const int mbase = wave * 16 + ((lane >> 4) << 3);

    for (int t = 0; t < T; ++t) {
        const unsigned short* hp = (t & 1) ? h1 : h0;
        unsigned short*       hn = (t & 1) ? h0 : h1;
        float* outT = out + (size_t)t * Bsz * U;

        v8f acc[8];
        #pragma unroll
        for (int n = 0; n < 8; ++n)
            acc[n] = v8f{0.f, 0.f, 0.f, 0.f, 0.f, 0.f, 0.f, 0.f};

        auto stage = [&](int buf, int kk) {
            const unsigned short* gA = hp + (size_t)lrow * U + kk + lhalf;
            unsigned short* dA = &sA[buf][lrow * 32 + lhalf];
            async_copy_b128(dA,     gA);
            async_copy_b128(dA + 8, gA + 8);
            const unsigned short* gB = WT + (size_t)(n0 + lrow) * U + kk + lhalf;
            unsigned short* dB = &sB[buf][lrow * 32 + lhalf];
            async_copy_b128(dB,     gB);
            async_copy_b128(dB + 8, gB + 8);
        };

        stage(0, 0);
        int cb = 0;
        for (int kk = 0; kk < U; kk += 32) {
            if (kk + 32 < U) {
                stage(cb ^ 1, kk + 32);
                wait_async<4>();
            } else {
                wait_async<0>();
            }
            __syncthreads();
            wmma_strip(sA[cb], sB[cb], wave, lr, kbA, kbB, acc);
            __syncthreads();
            cb ^= 1;
        }

        // epilogue: add xW (already in outT), tanh, write state + bf16 h
        #pragma unroll
        for (int nt = 0; nt < 8; ++nt) {
            const int u = n0 + nt * 16 + lr;
            #pragma unroll
            for (int i = 0; i < 8; ++i) {
                const int m = mbase + i;
                const size_t o = (size_t)m * U + u;
                float v = tanhf(acc[nt][i] + outT[o]);
                outT[o] = v;
                hn[o]   = f32_to_bf16(v);
            }
        }

        if (t + 1 < T)
            grid_sync(counter, gen, gridDim.x, (unsigned)(t + 1));
    }
}

// ---------------------------------------------------------------------------
// kernel_launch
//   ws: wxhT(2MB) | whhT(2MB) | h0(256KB) | h1(256KB) | sync(64B) |
//       [xbf 128MB if it fits]
// ---------------------------------------------------------------------------
extern "C" void kernel_launch(void* const* d_in, const int* in_sizes, int n_in,
                              void* d_out, int out_size, void* d_ws, size_t ws_size,
                              hipStream_t stream) {
    const float* X    = (const float*)d_in[0];
    const float* Wxh  = (const float*)d_in[1];
    const float* Whh  = (const float*)d_in[2];
    const float* bias = (const float*)d_in[3];
    float* out = (float*)d_out;

    unsigned short* wxhT = (unsigned short*)d_ws;
    unsigned short* whhT = wxhT + (size_t)D * U;
    unsigned short* h0   = whhT + (size_t)U * U;
    unsigned short* h1   = h0 + (size_t)Bsz * U;
    unsigned*       sync = (unsigned*)(h1 + (size_t)Bsz * U);
    unsigned short* xbf  = (unsigned short*)(sync + 16);

    const size_t base_need = ((size_t)2 * D * U + (size_t)2 * Bsz * U) * 2 + 64;
    const size_t x_elems   = (size_t)Bsz * T * D;
    const bool   preconv   = ws_size >= base_need + x_elems * 2;

    // 1) weight convert/transpose + zero h0 + zero barrier counters
    {
        int total = 2 * D * U + Bsz * U + 16;
        prep<<<(total + 255) / 256, 256, 0, stream>>>(Wxh, Whh, wxhT, whhT,
                                                      h0, sync);
    }

    // 2) input projection GEMM (+bias), scattered to [T,B,U]
    dim3 pg(U / 128, (Bsz * T) / 128);
    if (preconv) {
        xcvt<<<(unsigned)(x_elems / 16 / 256), 256, 0, stream>>>(X, xbf);
        xproj<true><<<pg, 256, 0, stream>>>(X, xbf, wxhT, bias, out);
    } else {
        xproj<false><<<pg, 256, 0, stream>>>(X, nullptr, wxhT, bias, out);
    }

    // 3) recurrent scan: single persistent kernel, grid barrier per step
    rnn_scan<<<U / 128, 256, 0, stream>>>(h0, h1, whhT, out,
                                          &sync[0], &sync[1]);
}
